// HDGT_encoder_31430570672764
// MI455X (gfx1250) — matrix operand order repaired and checked by reading, not compiled, not run on hardware
//
#include <hip/hip_runtime.h>

typedef __attribute__((ext_vector_type(16))) _Float16 v16h;
typedef __attribute__((ext_vector_type(8)))  float    v8f;

#define N_AGENT 4096
#define DIM     512
#define NHEAD   8
#define HDIM    64
#define E_OTHER 65536
#define E_L2A   65536
#define E_G2A   16384
#define INV_TEMP   0.125f               /* 1/sqrt(64) */
#define INV_LOG32  0.28853900817779268f /* 1/ln(32)  */

// ---------------------------------------------------------------------------
// monotonic float <-> uint keys for atomicMax-based segment max
// ---------------------------------------------------------------------------
__device__ __forceinline__ unsigned fkey(float x) {
  unsigned u = __float_as_uint(x);
  return (u & 0x80000000u) ? ~u : (u | 0x80000000u);
}
__device__ __forceinline__ float funkey(unsigned u) {
  return __uint_as_float((u & 0x80000000u) ? (u & 0x7fffffffu) : ~u);
}

// ---------------------------------------------------------------------------
// Weight pack: fp32 row-major [K][N] -> f16 B-fragment order.
// B-matrix 32x16 f16 order used here (contiguous k, not ISA table): lane holds col N = lane%16,
// element e holds K = e + 16*(lane>=16).  Packed index:
//   ((kt*ntiles + nt)*32 + lane)*16 + e   -> contiguous 32B per lane-fragment.
// ---------------------------------------------------------------------------
__global__ void k_pack(const float* __restrict__ W, _Float16* __restrict__ P,
                       int K, int N) {
  long long KN  = (long long)K * N;
  long long idx = (long long)blockIdx.x * blockDim.x + threadIdx.x;
  if (idx >= KN) return;
  long long base = (long long)blockIdx.y * KN;  // batch of matrices
  int e    = (int)(idx & 15);
  int lane = (int)((idx >> 4) & 31);
  long long tt = idx >> 9;
  int ntiles = N >> 4;
  int nt = (int)(tt % ntiles);
  int kt = (int)(tt / ntiles);
  int k = kt * 32 + ((lane >= 16) ? 16 : 0) + e;
  int n = nt * 16 + (lane & 15);
  P[base + idx] = (_Float16)W[base + (long long)k * N + n];
}

// ---------------------------------------------------------------------------
// Stage 32 rows of [Kdim] fp32 activations into LDS in A-fragment order
// (two 16-row M-tiles), with optional LayerNorm.  A-matrix 16x32 f16 layout
// (order used here, contiguous k): lane = grp*16 + row; lanes 0-15 hold K = {0..15},
// lanes 16-31 hold K = {16..31}.
// Block = 256 threads (8 lanes per row). Ends with __syncthreads().
// aF layout: [mtile(2)][ktiles][lane(32)][e(16)]
// ---------------------------------------------------------------------------
__device__ __forceinline__ void stage32(
    const float* __restrict__ src, long long row0, int Kdim,
    const float* __restrict__ gamma, const float* __restrict__ beta,
    _Float16* aF, float* red, float* stats) {
  const int tid = threadIdx.x;
  const int row = tid >> 3;   // 0..31
  const int sub = tid & 7;    // 0..7
  const float* xr = src + (row0 + row) * (long long)Kdim;
  float mu = 0.f, rs = 0.f;
  const bool doLN = (gamma != nullptr);
  if (doLN) {
    float s1 = 0.f, s2 = 0.f;
    for (int k = sub; k < Kdim; k += 8) { float v = xr[k]; s1 += v; s2 += v * v; }
    red[row * 8 + sub]       = s1;
    red[256 + row * 8 + sub] = s2;
    __syncthreads();
    if (sub == 0) {
      float a = 0.f, b = 0.f;
      for (int i = 0; i < 8; i++) { a += red[row * 8 + i]; b += red[256 + row * 8 + i]; }
      float m   = a / (float)Kdim;
      float var = b / (float)Kdim - m * m;
      stats[row]      = m;
      stats[32 + row] = rsqrtf(var + 1e-5f);
    }
    __syncthreads();
    mu = stats[row]; rs = stats[32 + row];
  }
  const int chunk  = Kdim >> 3;
  const int ktiles = Kdim >> 5;
  const int mt  = row >> 4;
  const int r16 = row & 15;
  for (int j = 0; j < chunk; j++) {
    int k = sub * chunk + j;
    float v = xr[k];
    if (doLN) v = (v - mu) * rs * gamma[k] + beta[k];
    int kt = k >> 5, kk = k & 31;
    int grp, e;
    grp = kk >> 4; e = kk & 15;
    aF[(((mt * ktiles + kt) * 32 + grp * 16 + r16) << 4) + e] = (_Float16)v;
  }
  __syncthreads();
}

// Two 16x16 output tiles sharing one B-fragment stream:
// K-loop of paired v_wmma_f32_16x16x32_f16 (2 WMMA per B load).
__device__ __forceinline__ void wave_gemm2(const _Float16* aF,
                                           const _Float16* __restrict__ Bp,
                                           int ktiles, int ntiles, int nt, int lane,
                                           v8f* cc) {
  v8f c0 = {};
  v8f c1 = {};
  for (int kt = 0; kt < ktiles; kt++) {
    v16h bv = *(const v16h*)(Bp + ((((long long)kt * ntiles + nt) * 32 + lane) << 4));
    v16h a0 = *(const v16h*)(aF + ((kt * 32 + lane) << 4));
    v16h a1 = *(const v16h*)(aF + (((ktiles + kt) * 32 + lane) << 4));
    c0 = __builtin_amdgcn_wmma_f32_16x16x32_f16(false, a0, false, bv,
                                                (short)0, c0, false, false);
    c1 = __builtin_amdgcn_wmma_f32_16x16x32_f16(false, a1, false, bv,
                                                (short)0, c1, false, false);
  }
  cc[0] = c0; cc[1] = c1;
}

// ---------------------------------------------------------------------------
// Generic tiled GEMM: out = act( [LN](src) @ W[t] + bias[t] ) [+ resid]
// grid.x = row tiles of 32, grid.y = type (predicated store) or 1.
// ---------------------------------------------------------------------------
template <int KDIM>
__global__ __launch_bounds__(256) void k_gemm(
    const float* __restrict__ src,
    const _Float16* __restrict__ Wp, long long wStride,
    float* __restrict__ outF, _Float16* __restrict__ outH, int outLd,
    const float* __restrict__ gamma, const float* __restrict__ beta, int gbStride,
    const float* __restrict__ bias, int biasStride,
    const float* __restrict__ resid,
    const int* __restrict__ types, const int* __restrict__ dstIdx,
    int Nout, int doRelu) {
  __shared__ __align__(32) _Float16 aF[KDIM * 32];
  __shared__ float red[512];
  __shared__ float stats[64];
  const int t = blockIdx.y;
  const long long row0 = (long long)blockIdx.x * 32;
  const float* g = gamma ? gamma + (long long)t * gbStride : nullptr;
  const float* b = beta  ? beta  + (long long)t * gbStride : nullptr;
  stage32(src, row0, KDIM, g, b, aF, red, stats);
  const _Float16* W = Wp + (long long)t * wStride;
  const int wave = threadIdx.x >> 5, lane = threadIdx.x & 31;
  const int ntiles = Nout >> 4, ktiles = KDIM >> 5;
  for (int nt = wave; nt < ntiles; nt += 8) {
    if (nt + 8 < ntiles)  // prefetch next B column-tile (global_prefetch_b8)
      __builtin_prefetch(W + (((long long)(nt + 8) * 32 + lane) << 4), 0, 1);
    v8f cc[2];
    wave_gemm2(aF, W, ktiles, ntiles, nt, lane, cc);
    const int col = nt * 16 + (lane & 15);
    const float bv = bias ? bias[(long long)t * biasStride + col] : 0.f;
#pragma unroll
    for (int mh = 0; mh < 2; mh++) {
#pragma unroll
      for (int r = 0; r < 8; r++) {
        long long row = row0 + mh * 16 + r + ((lane >> 4) << 3);
        if (types) {
          int rt = types[dstIdx ? dstIdx[row] : (int)row];
          if (rt != t) continue;
        }
        float v = cc[mh][r] + bv;
        if (doRelu) v = fmaxf(v, 0.f);
        if (resid) v += resid[row * (long long)outLd + col];
        if (outF) outF[row * (long long)outLd + col] = v;
        else      outH[row * (long long)outLd + col] = (_Float16)v;
      }
    }
  }
}

// ---------------------------------------------------------------------------
// Branch FCs: fs/fo/fl/fg = relu(x @ W[t,branch] + b) -> cat[:, branch*512+]
// grid = (N/32, 4 branches, 3 types)
// ---------------------------------------------------------------------------
__global__ __launch_bounds__(256) void k_branch(
    const float* __restrict__ selfH, const float* __restrict__ aout,
    const _Float16* __restrict__ pSFC, const _Float16* __restrict__ pAFC,
    const float* __restrict__ selfB, const float* __restrict__ attnB,
    const int* __restrict__ types, float* __restrict__ cat) {
  __shared__ __align__(32) _Float16 aF[DIM * 32];
  __shared__ float red[512];
  __shared__ float stats[64];
  const int br = blockIdx.y, t = blockIdx.z;
  const long long row0 = (long long)blockIdx.x * 32;
  const float* src = (br == 0) ? selfH : aout + (long long)(br - 1) * N_AGENT * DIM;
  stage32(src, row0, DIM, nullptr, nullptr, aF, red, stats);
  const _Float16* W = (br == 0) ? (pSFC + (long long)t * DIM * DIM)
                                : (pAFC + ((long long)t * 3 + (br - 1)) * DIM * DIM);
  const float* bias = (br == 0) ? (selfB + (long long)t * DIM)
                                : (attnB + ((long long)t * 3 + (br - 1)) * DIM);
  const int wave = threadIdx.x >> 5, lane = threadIdx.x & 31;
  for (int nt = wave; nt < 32; nt += 8) {
    v8f cc[2];
    wave_gemm2(aF, W, 16, 32, nt, lane, cc);
    int col = nt * 16 + (lane & 15);
    float bv = bias[col];
#pragma unroll
    for (int mh = 0; mh < 2; mh++) {
#pragma unroll
      for (int r = 0; r < 8; r++) {
        long long row = row0 + mh * 16 + r + ((lane >> 4) << 3);
        if (types[row] != t) continue;
        cat[row * 2048 + (long long)br * DIM + col] = fmaxf(cc[mh][r] + bv, 0.f);
      }
    }
  }
}

// ---------------------------------------------------------------------------
// FFN part 1: y = LN(x); h1 = silu(y@w1+b1) * (y@w3+b3)   (dual GEMM)
// ---------------------------------------------------------------------------
__global__ __launch_bounds__(256) void k_ffn1(
    const float* __restrict__ x,
    const _Float16* __restrict__ pW1, const _Float16* __restrict__ pW3,
    const float* __restrict__ gamma, const float* __restrict__ beta,
    const float* __restrict__ b1, const float* __restrict__ b3,
    const int* __restrict__ types, float* __restrict__ h1) {
  __shared__ __align__(32) _Float16 aF[DIM * 32];
  __shared__ float red[512];
  __shared__ float stats[64];
  const int t = blockIdx.y;
  const long long row0 = (long long)blockIdx.x * 32;
  stage32(x, row0, DIM, gamma + (long long)t * DIM, beta + (long long)t * DIM,
          aF, red, stats);
  const _Float16* W1 = pW1 + (long long)t * DIM * 2048;
  const _Float16* W3 = pW3 + (long long)t * DIM * 2048;
  const int wave = threadIdx.x >> 5, lane = threadIdx.x & 31;
  for (int nt = wave; nt < 128; nt += 8) {
    if (nt + 8 < 128)
      __builtin_prefetch(W1 + (((long long)(nt + 8) * 32 + lane) << 4), 0, 1);
    v8f cc1[2], cc3[2];
    wave_gemm2(aF, W1, 16, 128, nt, lane, cc1);
    wave_gemm2(aF, W3, 16, 128, nt, lane, cc3);
    int col = nt * 16 + (lane & 15);
    float bb1 = b1[(long long)t * 2048 + col];
    float bb3 = b3[(long long)t * 2048 + col];
#pragma unroll
    for (int mh = 0; mh < 2; mh++) {
#pragma unroll
      for (int r = 0; r < 8; r++) {
        long long row = row0 + mh * 16 + r + ((lane >> 4) << 3);
        if (types[row] != t) continue;
        float a   = cc1[mh][r] + bb1;
        float sil = a / (1.f + expf(-a));
        h1[row * 2048 + col] = sil * (cc3[mh][r] + bb3);
      }
    }
  }
}

// ---------------------------------------------------------------------------
// Edge attention (segment softmax with log(deg+1)/log32 rescale)
// ---------------------------------------------------------------------------
__global__ void k_zero(float* __restrict__ p, long long n) {
  long long i = (long long)blockIdx.x * blockDim.x + threadIdx.x;
  long long stride = (long long)gridDim.x * blockDim.x;
  for (; i < n; i += stride) p[i] = 0.f;
}

__global__ void k_deg(const int* __restrict__ dst, int E, float* __restrict__ deg) {
  int i = blockIdx.x * blockDim.x + threadIdx.x;
  if (i < E) atomicAdd(deg + dst[i], 1.f);
}

__global__ void k_score(const int* __restrict__ dst, int E,
                        const _Float16* __restrict__ kv, const float* __restrict__ q,
                        int chunk, const float* __restrict__ deg,
                        float* __restrict__ sbuf, unsigned* __restrict__ mkey) {
  int id = blockIdx.x * blockDim.x + threadIdx.x;
  if (id >= E * NHEAD) return;
  int e = id >> 3, h = id & 7;
  int d = dst[e];
  const float*    qp = q  + (long long)d * 1536 + h * 192 + chunk * 64;
  const _Float16* kp = kv + (long long)e * 1024 + h * 128;
  float s = 0.f;
  for (int i = 0; i < HDIM; i++) s += qp[i] * (float)kp[i];
  s *= INV_TEMP;
  s *= logf(deg[d] + 1.f) * INV_LOG32;
  sbuf[id] = s;
  atomicMax(mkey + d * NHEAD + h, fkey(s));
}

__global__ void k_expz(const int* __restrict__ dst, int E,
                       float* __restrict__ sbuf, const unsigned* __restrict__ mkey,
                       float* __restrict__ z) {
  int id = blockIdx.x * blockDim.x + threadIdx.x;
  if (id >= E * NHEAD) return;
  int e = id >> 3, h = id & 7;
  int d = dst[e];
  float ev = expf(sbuf[id] - funkey(mkey[d * NHEAD + h]));
  sbuf[id] = ev;
  atomicAdd(z + d * NHEAD + h, ev);
}

__global__ void k_accum(const int* __restrict__ dst, int E,
                        const float* __restrict__ sbuf, const float* __restrict__ z,
                        const _Float16* __restrict__ kv, float* __restrict__ aout) {
  int id = blockIdx.x * blockDim.x + threadIdx.x;
  if (id >= E * NHEAD) return;
  int e = id >> 3, h = id & 7;
  int d = dst[e];
  float a = sbuf[id] / fmaxf(z[d * NHEAD + h], 1e-9f);
  const _Float16* vp = kv + (long long)e * 1024 + h * 128 + 64;
  float* op = aout + (long long)d * DIM + h * HDIM;
  for (int i = 0; i < HDIM; i++) atomicAdd(op + i, a * (float)vp[i]);
}

// ---------------------------------------------------------------------------
// Host orchestration
// ---------------------------------------------------------------------------
extern "C" void kernel_launch(void* const* d_in, const int* in_sizes, int n_in,
                              void* d_out, int out_size, void* d_ws, size_t ws_size,
                              hipStream_t stream) {
  (void)in_sizes; (void)n_in; (void)out_size; (void)ws_size;
  const float* a_n_hidden = (const float*)d_in[0];
  const int*   a_n_type   = (const int*)  d_in[1];
  const float* self_e     = (const float*)d_in[2];
  const int*   other_dst  = (const int*)  d_in[3];
  const float* other_e    = (const float*)d_in[4];
  const int*   l2a_dst    = (const int*)  d_in[5];
  const float* l2a_e      = (const float*)d_in[6];
  const int*   g2a_dst    = (const int*)  d_in[7];
  const float* g2a_e      = (const float*)d_in[8];
  const float* wq_gamma   = (const float*)d_in[9];
  const float* wq_beta    = (const float*)d_in[10];
  const float* wq_W       = (const float*)d_in[11];
  const float* kvl_gamma  = (const float*)d_in[12];
  const float* kvl_beta   = (const float*)d_in[13];
  const float* kvl_W      = (const float*)d_in[14];
  const float* kvg_gamma  = (const float*)d_in[15];
  const float* kvg_beta   = (const float*)d_in[16];
  const float* kvg_W      = (const float*)d_in[17];
  const float* kvo_gamma  = (const float*)d_in[18];
  const float* kvo_beta   = (const float*)d_in[19];
  const float* kvo_W      = (const float*)d_in[20];
  const float* afc_W      = (const float*)d_in[21];
  const float* afc_b      = (const float*)d_in[22];
  const float* sfc_W      = (const float*)d_in[23];
  const float* sfc_b      = (const float*)d_in[24];
  const float* ofc_W      = (const float*)d_in[25];
  const float* ofc_b      = (const float*)d_in[26];
  const float* ffn_gamma  = (const float*)d_in[27];
  const float* ffn_beta   = (const float*)d_in[28];
  const float* ffn_w1     = (const float*)d_in[29];
  const float* ffn_b1     = (const float*)d_in[30];
  const float* ffn_w2     = (const float*)d_in[31];
  const float* ffn_b2     = (const float*)d_in[32];
  const float* ffn_w3     = (const float*)d_in[33];
  const float* ffn_b3     = (const float*)d_in[34];

  char* ws = (char*)d_ws;
  size_t off = 0;
  auto alloc = [&](size_t bytes) -> char* {
    char* p = ws + off;
    off += (bytes + 255) & ~(size_t)255;
    return p;
  };

  // packed f16 weights (B-fragment order)
  _Float16* pWQ  = (_Float16*)alloc(3ULL * 512 * 1536 * 2);
  _Float16* pKVL = (_Float16*)alloc(1ULL * 512 * 1024 * 2);
  _Float16* pKVG = (_Float16*)alloc(1ULL * 512 * 1024 * 2);
  _Float16* pKVO = (_Float16*)alloc(3ULL * 512 * 1024 * 2);
  _Float16* pAFC = (_Float16*)alloc(9ULL * 512 * 512 * 2);
  _Float16* pSFC = (_Float16*)alloc(3ULL * 512 * 512 * 2);
  _Float16* pOFC = (_Float16*)alloc(3ULL * 2048 * 512 * 2);
  _Float16* pW1  = (_Float16*)alloc(3ULL * 512 * 2048 * 2);
  _Float16* pW3  = (_Float16*)alloc(3ULL * 512 * 2048 * 2);
  _Float16* pW2  = (_Float16*)alloc(3ULL * 2048 * 512 * 2);
  // activations
  float*    qBuf = (float*)   alloc((size_t)N_AGENT * 1536 * 4);
  _Float16* kvL  = (_Float16*)alloc((size_t)E_L2A * 1024 * 2);
  _Float16* kvG  = (_Float16*)alloc((size_t)E_G2A * 1024 * 2);
  _Float16* kvO  = (_Float16*)alloc((size_t)E_OTHER * 1024 * 2);
  // contiguous zero-init region: deg | mkey | z | attn_out
  const long long DEGN = 3LL * N_AGENT;
  const long long MKN  = 3LL * N_AGENT * NHEAD;
  const long long ZN   = MKN;
  const long long AON  = 3LL * N_AGENT * DIM;
  float*    zeroBase = (float*)alloc((size_t)(DEGN + MKN + ZN + AON) * 4);
  float*    deg  = zeroBase;
  unsigned* mkey = (unsigned*)(zeroBase + DEGN);
  float*    zsum = zeroBase + DEGN + MKN;
  float*    aout = zeroBase + DEGN + MKN + ZN;
  float* sO   = (float*)alloc((size_t)E_OTHER * NHEAD * 4);
  float* sL   = (float*)alloc((size_t)E_L2A * NHEAD * 4);
  float* sG   = (float*)alloc((size_t)E_G2A * NHEAD * 4);
  float* cat  = (float*)alloc((size_t)N_AGENT * 2048 * 4);
  float* out1 = (float*)alloc((size_t)N_AGENT * 512 * 4);
  float* h1   = (float*)alloc((size_t)N_AGENT * 2048 * 4);

  auto pack = [&](const float* W, _Float16* P, int K, int N, int B) {
    long long KN = (long long)K * N;
    dim3 g((unsigned)((KN + 255) / 256), B);
    k_pack<<<g, 256, 0, stream>>>(W, P, K, N);
  };
  pack(wq_W,   pWQ,  512, 1536, 3);
  pack(kvl_W,  pKVL, 512, 1024, 1);
  pack(kvg_W,  pKVG, 512, 1024, 1);
  pack(kvo_W,  pKVO, 512, 1024, 3);
  pack(afc_W,  pAFC, 512, 512,  9);
  pack(sfc_W,  pSFC, 512, 512,  3);
  pack(ofc_W,  pOFC, 2048, 512, 3);
  pack(ffn_w1, pW1,  512, 2048, 3);
  pack(ffn_w3, pW3,  512, 2048, 3);
  pack(ffn_w2, pW2,  2048, 512, 3);

  k_zero<<<2048, 256, 0, stream>>>(zeroBase, DEGN + MKN + ZN + AON);

  // Q projection: LN(a_n_hidden; wq[t]) @ wq_W[t]  -> qBuf [N][1536]
  k_gemm<512><<<dim3(N_AGENT / 32, 3), 256, 0, stream>>>(
      a_n_hidden, pWQ, 512LL * 1536, qBuf, nullptr, 1536,
      wq_gamma, wq_beta, 512, nullptr, 0, nullptr, a_n_type, nullptr, 1536, 0);
  // K/V projections -> f16 [E][1024]
  k_gemm<512><<<dim3(E_L2A / 32, 1), 256, 0, stream>>>(
      l2a_e, pKVL, 0, nullptr, kvL, 1024,
      kvl_gamma, kvl_beta, 0, nullptr, 0, nullptr, nullptr, nullptr, 1024, 0);
  k_gemm<512><<<dim3(E_G2A / 32, 1), 256, 0, stream>>>(
      g2a_e, pKVG, 0, nullptr, kvG, 1024,
      kvg_gamma, kvg_beta, 0, nullptr, 0, nullptr, nullptr, nullptr, 1024, 0);
  k_gemm<512><<<dim3(E_OTHER / 32, 3), 256, 0, stream>>>(
      other_e, pKVO, 512LL * 1024, nullptr, kvO, 1024,
      kvo_gamma, kvo_beta, 512, nullptr, 0, nullptr, a_n_type, other_dst, 1024, 0);

  // segment-softmax attention per edge type (0=other,1=lane,2=poly)
  auto attn = [&](const int* dst, int E, const _Float16* kv, int chunk, int et,
                  float* sbuf) {
    float*    dg = deg  + (long long)et * N_AGENT;
    unsigned* mk = mkey + (long long)et * N_AGENT * NHEAD;
    float*    zz = zsum + (long long)et * N_AGENT * NHEAD;
    float*    ao = aout + (long long)et * N_AGENT * DIM;
    k_deg  <<<(E + 255) / 256, 256, 0, stream>>>(dst, E, dg);
    int gEH = (E * NHEAD + 255) / 256;
    k_score<<<gEH, 256, 0, stream>>>(dst, E, kv, qBuf, chunk, dg, sbuf, mk);
    k_expz <<<gEH, 256, 0, stream>>>(dst, E, sbuf, mk, zz);
    k_accum<<<gEH, 256, 0, stream>>>(dst, E, sbuf, zz, kv, ao);
  };
  attn(other_dst, E_OTHER, kvO, 0, 0, sO);
  attn(l2a_dst,   E_L2A,   kvL, 1, 1, sL);
  attn(g2a_dst,   E_G2A,   kvG, 2, 2, sG);

  // branch FCs -> cat [N][2048] (order: self, other, lane, poly)
  k_branch<<<dim3(N_AGENT / 32, 4, 3), 256, 0, stream>>>(
      self_e, aout, pSFC, pAFC, sfc_b, afc_b, a_n_type, cat);

  // out_fc: cat @ W[t] + b + a_n_hidden -> out1
  k_gemm<2048><<<dim3(N_AGENT / 32, 3), 256, 0, stream>>>(
      cat, pOFC, 2048LL * 512, out1, nullptr, 512,
      nullptr, nullptr, 0, ofc_b, 512, a_n_hidden, a_n_type, nullptr, 512, 0);

  // FFN
  k_ffn1<<<dim3(N_AGENT / 32, 3), 256, 0, stream>>>(
      out1, pW1, pW3, ffn_gamma, ffn_beta, ffn_b1, ffn_b3, a_n_type, h1);
  k_gemm<2048><<<dim3(N_AGENT / 32, 3), 256, 0, stream>>>(
      h1, pW2, 2048LL * 512, (float*)d_out, nullptr, 512,
      nullptr, nullptr, 0, ffn_b2, 512, out1, a_n_type, nullptr, 512, 0);
}
